// LSTM_74328704025085
// MI455X (gfx1250) — matrix operand order repaired and checked by reading, not compiled, not run on hardware
//
#include <hip/hip_runtime.h>
#include <math.h>

// Problem constants (from reference)
#define S_   512
#define B_   64
#define I_   768
#define H_   768
#define KTOT 1536                 // I + H
#define BH   (B_ * H_)            // 49152
#define HALL_N ((S_ + 1) * BH)    // 25214976  (S+1, B, H)
#define SBH  ((size_t)S_ * BH)    // 25165824  (S, B, H)

typedef _Float16 f16;
typedef __attribute__((ext_vector_type(16))) _Float16 v16h;
typedef __attribute__((ext_vector_type(8)))  float    v8f;

// LDS B-tile geometry (phase 1): 64 cols x 32 K-halves, padded column stride
#define COL_BYTES   80            // 64B of K data + 16B pad -> conflict-free b128 reads
#define BUF_BYTES   (64 * COL_BYTES)   // 5120 B per buffer
#define NKSTEPS     (I_ / 32)     // 24

// ---------------------------------------------------------------------------
// CDNA5 helpers
// ---------------------------------------------------------------------------
__device__ __forceinline__ v8f wmma_f32_f16(v16h a, v16h b, v8f c) {
  // (neg_a, A, neg_b, B, c_mod, C, reuse_a, reuse_b)
  return __builtin_amdgcn_wmma_f32_16x16x32_f16(false, a, false, b, (short)0, c,
                                                false, false);
}

// Async DMA: global -> LDS, 16 bytes per lane, tracked by ASYNCcnt.
__device__ __forceinline__ void async_load_b128(unsigned lds_addr,
                                                const void* gaddr) {
  asm volatile("global_load_async_to_lds_b128 %0, %1, off"
               :
               : "v"(lds_addr), "v"((unsigned long long)(uintptr_t)gaddr)
               : "memory");
}

template <int N>
__device__ __forceinline__ void wait_async() {
#if __has_builtin(__builtin_amdgcn_s_wait_asynccnt)
  __builtin_amdgcn_s_wait_asynccnt((short)N);
#else
  asm volatile("s_wait_asynccnt %0" :: "i"(N) : "memory");
#endif
}

// Flat LDS pointer -> LDS byte offset (aperture rule: offset is the low 32 bits of the flat address).
__device__ __forceinline__ unsigned lds_off(const void* p) {
  return (unsigned)(uintptr_t)p;
}

// A fragment, 16x32 f16, built from f32 row data.
// Lane L: m = L&15, kh = L>>4.  Half h in [0,16): K = 16*kh + h.
__device__ __forceinline__ v16h load_a_frag_f32(const float* __restrict__ p, int kh) {
  const float* q = p + 16 * kh;
  float4 r0 = *(const float4*)(q);
  float4 r1 = *(const float4*)(q + 4);
  float4 r2 = *(const float4*)(q + 8);
  float4 r3 = *(const float4*)(q + 12);
  v16h a;
  a[0]  = (f16)r0.x; a[1]  = (f16)r0.y; a[2]  = (f16)r0.z; a[3]  = (f16)r0.w;
  a[4]  = (f16)r1.x; a[5]  = (f16)r1.y; a[6]  = (f16)r1.z; a[7]  = (f16)r1.w;
  a[8]  = (f16)r2.x; a[9]  = (f16)r2.y; a[10] = (f16)r2.z; a[11] = (f16)r2.w;
  a[12] = (f16)r3.x; a[13] = (f16)r3.y; a[14] = (f16)r3.z; a[15] = (f16)r3.w;
  return a;
}

// B fragment from N-major f16 weights in global memory (phase 2 path):
// Lane L: n = col, kh = L>>4; half h: K = 16*kh + h -> 16 contiguous halves.
__device__ __forceinline__ v16h load_b_frag(const f16* __restrict__ col_k0, int kh) {
  return *(const v16h*)(col_k0 + 16 * kh);
}

__device__ __forceinline__ float sigmoidf_(float x) {
  return 1.0f / (1.0f + __expf(-x));
}

// ---------------------------------------------------------------------------
// Phase 0a: convert + transpose weights to f16, N-major:  Wt[g][n][k] = W_g[k][n]
// ---------------------------------------------------------------------------
__global__ void cvt_transpose_w(const float* __restrict__ Wf,
                                const float* __restrict__ Wc,
                                const float* __restrict__ Wo,
                                f16* __restrict__ Wt) {
  int idx = blockIdx.x * blockDim.x + threadIdx.x;
  const int per = KTOT * H_;  // 1179648
  if (idx >= 3 * per) return;
  int g = idx / per;
  int r = (idx - g * per) / H_;   // k  in [0, 1536)
  int c = idx % H_;               // n  in [0, 768)
  const float* W = (g == 0) ? Wf : ((g == 1) ? Wc : Wo);
  Wt[(size_t)g * per + (size_t)c * KTOT + r] = (f16)W[(size_t)r * H_ + c];
}

// ---------------------------------------------------------------------------
// Phase 0b: zero initial state h0, C0 (d_out is poisoned by the harness)
// ---------------------------------------------------------------------------
__global__ void init_state(float* __restrict__ out) {
  int idx = blockIdx.x * blockDim.x + threadIdx.x;
  if (idx < BH) {
    out[idx] = 0.0f;                       // Hall[0]
    out[(size_t)HALL_N + idx] = 0.0f;      // Call[0]
  }
}

// ---------------------------------------------------------------------------
// Phase 1: x-projections.  Gx_g = X @ Wx_g + b_g  for g in {f,c,o}.
// One wave computes a 16x64 tile; the whole workgroup (8 waves) shares one
// 64-column B tile staged in LDS via GLOBAL_LOAD_ASYNC_TO_LDS_B128,
// double-buffered so the async DMA of K-step i+1 overlaps WMMAs on step i.
// grid = (256, 12, 3), block = 256 (8 waves).
// ---------------------------------------------------------------------------
__global__ __launch_bounds__(256) void xproj_gemm(
    const float* __restrict__ X, const f16* __restrict__ Wt,
    const float* __restrict__ bf, const float* __restrict__ bc,
    const float* __restrict__ bo, float* __restrict__ out) {
  __shared__ __align__(16) unsigned char smem[2 * BUF_BYTES];

  const int tid  = threadIdx.x;
  const int lane = tid & 31;
  const int wave = tid >> 5;
  const int li = lane & 15;
  const int kh = lane >> 4;

  const int g  = blockIdx.z;
  const int m0 = (blockIdx.x * 8 + wave) * 16;   // row tile (0..32752)
  const int n0 = blockIdx.y * 64;                // col tile

  const float* bias = (g == 0) ? bf : ((g == 1) ? bc : bo);
  float* og = out + (size_t)2 * HALL_N + (size_t)g * SBH;  // Fs / Is / Os region

  const f16* wg = Wt + (size_t)g * KTOT * H_;

  // Async copy assignment: thread -> (column, 16B quarter of the 64B K-run)
  const int lcol = tid >> 2;   // 0..63
  const int lq   = tid & 3;    // 0..3
  const f16* gsrc = wg + (size_t)(n0 + lcol) * KTOT + I_ + lq * 8;  // x-part k>=768
  const unsigned ldst = lds_off(smem) + lcol * COL_BYTES + lq * 16;

  const float* xrow = X + (size_t)(m0 + li) * I_;

  v8f acc0 = {}, acc1 = {}, acc2 = {}, acc3 = {};

  // Prologue: fill buffer 0
  async_load_b128(ldst, gsrc);

#pragma unroll 1
  for (int i = 0; i < NKSTEPS; ++i) {
    const int k0 = i * 32;
    const int cur = i & 1;
    if (i + 1 < NKSTEPS) {
      async_load_b128(ldst + (1 - cur) * BUF_BYTES, gsrc + (i + 1) * 32);
      wait_async<1>();   // oldest batch (this iter's buffer) has landed
    } else {
      wait_async<0>();
    }
    __syncthreads();     // tile visible to all 8 waves

    const unsigned char* lb = smem + cur * BUF_BYTES + kh * 32;
    v16h a  = load_a_frag_f32(xrow + k0, kh);
    v16h b0 = *(const v16h*)(lb + (0 * 16 + li) * COL_BYTES);
    v16h b1 = *(const v16h*)(lb + (1 * 16 + li) * COL_BYTES);
    v16h b2 = *(const v16h*)(lb + (2 * 16 + li) * COL_BYTES);
    v16h b3 = *(const v16h*)(lb + (3 * 16 + li) * COL_BYTES);
    acc0 = wmma_f32_f16(a, b0, acc0);
    acc1 = wmma_f32_f16(a, b1, acc1);
    acc2 = wmma_f32_f16(a, b2, acc2);
    acc3 = wmma_f32_f16(a, b3, acc3);

    __syncthreads();     // all reads done before this buffer is refilled
  }

  // bias + store; C/D layout: lane li = n, VGPR j = row (j + 8*kh)
  v8f accs[4] = {acc0, acc1, acc2, acc3};
#pragma unroll
  for (int t = 0; t < 4; ++t) {
    const int n = n0 + t * 16 + li;
    const float bv = bias[n];
#pragma unroll
    for (int j = 0; j < 8; ++j) {
      const int m = m0 + j + 8 * kh;
      og[(size_t)m * H_ + n] = accs[t][j] + bv;
    }
  }
}

// ---------------------------------------------------------------------------
// Phase 2: one recurrent step.  Gh_g = h_{s-1} @ Wh_g (64x768x768, 3 gates),
// fused with gate nonlinearities, cell/hidden update, and all output writes.
// One wave: 16 rows x 32 cols x 3 gates (6 accumulators). 96 waves = 12 blocks.
// Direct global B loads (weights are L2-resident; this phase is latency-bound).
// ---------------------------------------------------------------------------
__global__ __launch_bounds__(256) void lstm_step(const f16* __restrict__ Wt,
                                                 float* __restrict__ out,
                                                 int s) {
  const int lane = threadIdx.x & 31;
  const int wave = threadIdx.x >> 5;
  const int li = lane & 15;
  const int kh = lane >> 4;

  const int gw = blockIdx.x * 8 + wave;   // 0..95
  const int m0 = (gw & 3) * 16;           // batch-row tile (B=64 -> 4 tiles)
  const int n0 = (gw >> 2) * 32;          // col tile (768 -> 24 tiles)

  float* Hall = out;
  float* Call = out + (size_t)HALL_N;
  float* Fs   = out + (size_t)2 * HALL_N;
  float* Is   = Fs + SBH;
  float* Os   = Is + SBH;

  const size_t sBH = (size_t)s * BH;
  const float* hrow = Hall + sBH + (size_t)(m0 + li) * H_;  // h_{s-1}

  // B column bases: h-part of W is k in [0,768)
  const f16* cb[3][2];
#pragma unroll
  for (int g = 0; g < 3; ++g) {
#pragma unroll
    for (int t = 0; t < 2; ++t)
      cb[g][t] = Wt + (size_t)g * KTOT * H_ + (size_t)(n0 + t * 16 + li) * KTOT;
  }

  v8f acc[3][2];
#pragma unroll
  for (int g = 0; g < 3; ++g)
#pragma unroll
    for (int t = 0; t < 2; ++t) acc[g][t] = (v8f){};

#pragma unroll 2
  for (int k0 = 0; k0 < H_; k0 += 32) {
    v16h a = load_a_frag_f32(hrow + k0, kh);
#pragma unroll
    for (int g = 0; g < 3; ++g) {
#pragma unroll
      for (int t = 0; t < 2; ++t) {
        v16h b = load_b_frag(cb[g][t] + k0, kh);
        acc[g][t] = wmma_f32_f16(a, b, acc[g][t]);
      }
    }
  }

  // Fused gates + state update.  Gx (with bias) was staged in Fs/Is/Os by phase 1.
#pragma unroll
  for (int t = 0; t < 2; ++t) {
    const int n = n0 + t * 16 + li;
#pragma unroll
    for (int j = 0; j < 8; ++j) {
      const int m = m0 + j + 8 * kh;
      const size_t idx = (size_t)m * H_ + n;
      const float gxf = Fs[sBH + idx];
      const float gxc = Is[sBH + idx];
      const float gxo = Os[sBH + idx];
      const float cprev = Call[sBH + idx];

      const float fv = sigmoidf_(acc[0][t][j] + gxf);   // f (and i == f per source)
      const float ct = tanhf(acc[1][t][j] + gxc);
      const float ov = sigmoidf_(acc[2][t][j] + gxo);
      const float cn = fv * (cprev + ct);               // f*C + i*tanh(c~), i == f
      const float hn = ov * tanhf(cn);

      Fs[sBH + idx] = fv;
      Is[sBH + idx] = fv;
      Os[sBH + idx] = ov;
      Call[sBH + BH + idx] = cn;   // Call[s+1]
      Hall[sBH + BH + idx] = hn;   // Hall[s+1]
    }
  }
}

// ---------------------------------------------------------------------------
// Host-side launcher
// ---------------------------------------------------------------------------
extern "C" void kernel_launch(void* const* d_in, const int* in_sizes, int n_in,
                              void* d_out, int out_size, void* d_ws, size_t ws_size,
                              hipStream_t stream) {
  (void)in_sizes; (void)n_in; (void)out_size; (void)ws_size;
  const float* x  = (const float*)d_in[0];
  const float* Wf = (const float*)d_in[1];
  const float* bf = (const float*)d_in[2];
  const float* Wc = (const float*)d_in[3];
  const float* bc = (const float*)d_in[4];
  const float* Wo = (const float*)d_in[5];
  const float* bo = (const float*)d_in[6];
  float* out = (float*)d_out;
  f16* Wt = (f16*)d_ws;   // 3 * 1536 * 768 halves = ~7.1 MB of workspace

  // Phase 0: weight convert/transpose + zero initial state
  const int wtot = 3 * KTOT * H_;
  cvt_transpose_w<<<(wtot + 255) / 256, 256, 0, stream>>>(Wf, Wc, Wo, Wt);
  init_state<<<(BH + 255) / 256, 256, 0, stream>>>(out);

  // Phase 1: x projections for all timesteps (parallel, WMMA + async LDS DMA)
  dim3 g1(256, 12, 3);
  xproj_gemm<<<g1, 256, 0, stream>>>(x, Wt, bf, bc, bo, out);

  // Phase 2: sequential recurrence, one launch per timestep
  for (int s = 0; s < S_; ++s) {
    lstm_step<<<12, 256, 0, stream>>>(Wt, out, s);
  }
}
